// Toy_54812372631591
// MI455X (gfx1250) — matrix operand order repaired and checked by reading, not compiled, not run on hardware
//
#include <hip/hip_runtime.h>
#include <hip/hip_bf16.h>

#define VOCAB  50000
#define VDIM   256
#define TOPICS 512
#define KDIM   256

typedef __bf16 bf16_t;
typedef __attribute__((ext_vector_type(16))) __bf16       v16bf;
typedef __attribute__((ext_vector_type(2)))  __bf16       v2bf;
typedef __attribute__((ext_vector_type(8)))  float        v8f;
typedef __attribute__((ext_vector_type(4)))  float        f32x4;
typedef __attribute__((ext_vector_type(2)))  float        f32x2;
typedef __attribute__((ext_vector_type(4)))  unsigned int u32x4;

union Frag32 {
  v16bf bf;
  u32x4 q[2];
};

// native hardware converts (v_cvt_bf16_f32 / v_cvt_pk_bf16_f32), RNE
__device__ __forceinline__ bf16_t f2bf(float f) { return (bf16_t)f; }
__device__ __forceinline__ float  bf2f(bf16_t b) { return (float)b; }

// A-operand (16x32 bf16) from LDS: lane l holds row l%16; 16 contiguous K
// starting at half*16 (two 16B ds_load_b128 per fragment).
__device__ __forceinline__ v16bf load_a_lds(const bf16_t* base, int strideElems, int kc) {
  int lane = threadIdx.x & 31;
  int row  = lane & 15;
  int half = lane >> 4;
  const bf16_t* p0 = base + row * strideElems + kc * 32 + half * 16;
  Frag32 f;
  f.q[0] = *(const u32x4*)p0;
  f.q[1] = *(const u32x4*)(p0 + 8);
  return f.bf;
}

// B-operand (32x16 bf16) from global: lane l = column col0 + l%16,
// 16 contiguous K values (32B) starting at kc*32 + half*16.
__device__ __forceinline__ v16bf load_b_glb(const bf16_t* base, int strideElems, int col0, int kc) {
  int lane = threadIdx.x & 31;
  int n    = lane & 15;
  int half = lane >> 4;
  const bf16_t* p = base + (size_t)(col0 + n) * strideElems + kc * 32 + half * 16;
  Frag32 f;
  f.q[0] = *(const u32x4*)p;
  f.q[1] = *(const u32x4*)(p + 8);
  return f.bf;
}

__device__ __forceinline__ v8f wmma_bf16(v16bf a, v16bf b, v8f c) {
  return __builtin_amdgcn_wmma_f32_16x16x32_bf16(false, a, false, b, (short)0, c, false, false);
}

__device__ __forceinline__ float waveRedSum(float v) {
  #pragma unroll
  for (int m = 16; m >= 1; m >>= 1) v += __shfl_xor(v, m, 32);
  return v;
}
__device__ __forceinline__ float waveRedMax(float v) {
  #pragma unroll
  for (int m = 16; m >= 1; m >>= 1) v = fmaxf(v, __shfl_xor(v, m, 32));
  return v;
}
// reduce within each 16-lane half (masks 1..8 never cross lane 15/16 boundary)
__device__ __forceinline__ float halfRedSum(float v) {
  #pragma unroll
  for (int m = 8; m >= 1; m >>= 1) v += __shfl_xor(v, m, 32);
  return v;
}

// ---- per-wave LDS slice layout (bytes) ----
#define SSTRIDE 516          // f32 elems/row: 516%64==4 dwords -> conflict-free transposed reads
#define ASTRIDE 520          // bf16 elems/row: 1040B, 16B multiple (A-frag b128 alignment)
#define WBUF_STRIDE 264      // bf16 elems/row: 528B, 16B multiple
#define SBUF_OFF    0        // 16*516*4 = 33024 : s / alpha (f32)
#define ABUF_OFF    33024    // 16*520*2 = 16640 : alpha bf16
#define WBUF_OFF    49664    // 16*264*2 = 8448  : w2v tile bf16
#define CBUF_OFF    58112    // 16*264*2 = 8448  : wB then ctx bf16
#define WAVE_LDS_BYTES 66560

extern "C" __global__ void __launch_bounds__(128, 1)
topic_fused_kernel(const float* __restrict__ w2v,
                   const float* __restrict__ sigma,
                   const bf16_t* __restrict__ Bt,
                   const bf16_t* __restrict__ Abf,
                   const bf16_t* __restrict__ t2vbf,
                   const bf16_t* __restrict__ t2vT,
                   float* __restrict__ alpha_out,
                   float* __restrict__ s_out,
                   float* __restrict__ P_out,
                   float* __restrict__ tsums,
                   float* __restrict__ sumsq)
{
  extern __shared__ char lds[];
  const int lane = threadIdx.x & 31;
  const int wave = threadIdx.x >> 5;
  const int v0   = blockIdx.x * 64 + wave * 16;   // 16 vocab rows per wave

  // VOCAB % 16 == 0 -> each wave's strip is entirely valid or entirely
  // invalid; wave-uniform early-out keeps EXEC all-ones for every WMMA.
  if (v0 >= VOCAB) return;

  char*   wl   = lds + wave * WAVE_LDS_BYTES;
  float*  sbuf = (float*)(wl + SBUF_OFF);
  bf16_t* abuf = (bf16_t*)(wl + ABUF_OFF);
  bf16_t* wbuf = (bf16_t*)(wl + WBUF_OFF);
  bf16_t* cbuf = (bf16_t*)(wl + CBUF_OFF);

  const float inv_sigma = 1.0f / sigma[0];

  // ---- phase 1: stage 16x256 w2v tile as bf16 (b64 loads, packed cvt) ----
  for (int i = 0; i < 64; ++i) {
    int idx = i * 32 + lane;          // pair index
    int row = idx >> 7;               // 128 pairs per row
    int col = (idx & 127) * 2;
    f32x2 w = *(const f32x2*)(w2v + (size_t)(v0 + row) * VDIM + col);
    v2bf b;
    b.x = f2bf(w.x);
    b.y = f2bf(w.y);
    *(v2bf*)(wbuf + row * WBUF_STRIDE + col) = b;
  }

  const int nCol  = lane & 15;
  const int mBase = (lane >> 4) * 8;

  // ---- phase 2: wB = w2v_tile @ B   (M=16, N=256, K=256) -> cbuf ----
  for (int nt = 0; nt < 16; ++nt) {
    v8f acc = {};
    #pragma unroll
    for (int kc = 0; kc < 8; ++kc) {
      v16bf a = load_a_lds(wbuf, WBUF_STRIDE, kc);
      v16bf b = load_b_glb(Bt, VDIM, nt * 16, kc);
      acc = wmma_bf16(a, b, acc);
    }
    #pragma unroll
    for (int r = 0; r < 8; ++r)
      cbuf[(mBase + r) * WBUF_STRIDE + nt * 16 + nCol] = f2bf(acc[r]);
  }

  // ---- phase 3: s = wB @ t2v^T   (M=16, N=512, K=256) -> sbuf f32 ----
  for (int nt = 0; nt < 32; ++nt) {
    v8f acc = {};
    #pragma unroll
    for (int kc = 0; kc < 8; ++kc) {
      v16bf a = load_a_lds(cbuf, WBUF_STRIDE, kc);
      v16bf b = load_b_glb(t2vbf, KDIM, nt * 16, kc);
      acc = wmma_bf16(a, b, acc);
    }
    #pragma unroll
    for (int r = 0; r < 8; ++r)
      sbuf[(mBase + r) * SSTRIDE + nt * 16 + nCol] = acc[r];
  }

  // ---- store s, transposed through LDS (s_out only 4B-aligned -> b32) ----
  {
    int m = lane & 15, half = lane >> 4;
    size_t vcol = (size_t)(v0 + m);
    for (int tt = 0; tt < 256; ++tt) {
      int t = tt * 2 + half;
      s_out[(size_t)t * VOCAB + vcol] = sbuf[m * SSTRIDE + t];
    }
  }

  // ---- phase 4: softmax over topics per row; RL statistics ----
  float tacc[16];
  #pragma unroll
  for (int j = 0; j < 16; ++j) tacc[j] = 0.0f;
  float sq = 0.0f;

  for (int m = 0; m < 16; ++m) {
    float* srow = sbuf + m * SSTRIDE;
    float mx = -3.402823466e38f;
    #pragma unroll
    for (int j = 0; j < 16; ++j) mx = fmaxf(mx, srow[lane + j * 32]);
    mx = waveRedMax(mx);
    float sum = 0.0f;
    #pragma unroll
    for (int j = 0; j < 16; ++j) {
      float e = __expf(srow[lane + j * 32] - mx);
      srow[lane + j * 32] = e;
      sum += e;
    }
    sum = waveRedSum(sum);
    float inv = 1.0f / sum;
    #pragma unroll
    for (int j = 0; j < 16; ++j) {
      float a = srow[lane + j * 32] * inv;
      srow[lane + j * 32] = a;                         // alpha f32 (for store)
      abuf[m * ASTRIDE + lane + j * 32] = f2bf(a);     // alpha bf16 (for GEMM)
      sq += a * a;
      tacc[j] += a;
    }
  }
  #pragma unroll
  for (int j = 0; j < 16; ++j)
    atomicAdd(&tsums[lane + j * 32], tacc[j]);
  sq = waveRedSum(sq);
  if (lane == 0) atomicAdd(sumsq, sq);

  // ---- store alpha: transposed, b128 per lane (16B-aligned: VOCAB%4==0) ----
  {
    int t8 = lane >> 2;             // 8 topics per iteration
    int vi = (lane & 3) * 4;        // 4 consecutive vocab columns per lane
    for (int tt = 0; tt < 64; ++tt) {
      int t = tt * 8 + t8;
      f32x4 val;
      val.x = sbuf[(vi + 0) * SSTRIDE + t];
      val.y = sbuf[(vi + 1) * SSTRIDE + t];
      val.z = sbuf[(vi + 2) * SSTRIDE + t];
      val.w = sbuf[(vi + 3) * SSTRIDE + t];
      *(f32x4*)(alpha_out + (size_t)t * VOCAB + v0 + vi) = val;
    }
  }

  // ---- phase 5: ctx_v = alpha @ t2v  (M=16, N=256, K=512) -> cbuf ----
  for (int nt = 0; nt < 16; ++nt) {
    v8f acc = {};
    #pragma unroll
    for (int kc = 0; kc < 16; ++kc) {
      v16bf a = load_a_lds(abuf, ASTRIDE, kc);
      v16bf b = load_b_glb(t2vT, TOPICS, nt * 16, kc);
      acc = wmma_bf16(a, b, acc);
    }
    #pragma unroll
    for (int r = 0; r < 8; ++r)
      cbuf[(mBase + r) * WBUF_STRIDE + nt * 16 + nCol] = f2bf(acc[r]);
  }

  // ---- phase 6: mu = ctx_v @ A^T (K=256); fuse P = (1/s)*sum(diff^2) ----
  float pacc[8];
  #pragma unroll
  for (int r = 0; r < 8; ++r) pacc[r] = 0.0f;

  for (int nt = 0; nt < 16; ++nt) {
    v8f acc = {};
    #pragma unroll
    for (int kc = 0; kc < 8; ++kc) {
      v16bf a = load_a_lds(cbuf, WBUF_STRIDE, kc);
      v16bf b = load_b_glb(Abf, KDIM, nt * 16, kc);
      acc = wmma_bf16(a, b, acc);
    }
    int n = nt * 16 + nCol;
    #pragma unroll
    for (int r = 0; r < 8; ++r) {
      float w = bf2f(wbuf[(mBase + r) * WBUF_STRIDE + n]);
      float d = w - acc[r];
      pacc[r] += d * d;
    }
  }
  #pragma unroll
  for (int r = 0; r < 8; ++r) pacc[r] = halfRedSum(pacc[r]);
  if (nCol == 0) {
    #pragma unroll
    for (int r = 0; r < 8; ++r)
      P_out[v0 + mBase + r] = pacc[r] * inv_sigma;
  }
}

// ---- bf16 operand staging: B^T, A, t2v, t2v^T (all K-contiguous for WMMA B-op) ----
extern "C" __global__ void topic_prep_kernel(const float* __restrict__ B,
                                             const float* __restrict__ A,
                                             const float* __restrict__ t2v,
                                             bf16_t* __restrict__ Bt,
                                             bf16_t* __restrict__ Abf,
                                             bf16_t* __restrict__ t2vbf,
                                             bf16_t* __restrict__ t2vT)
{
  int i = blockIdx.x * blockDim.x + threadIdx.x;
  if (i < 65536) {                       // Bt[n][k] = B[k][n]
    int n = i >> 8, k = i & 255;
    Bt[i] = f2bf(B[k * 256 + n]);
  } else if (i < 131072) {               // A row-major copy
    int j = i - 65536;
    Abf[j] = f2bf(A[j]);
  } else if (i < 262144) {               // t2v row-major copy
    int j = i - 131072;
    t2vbf[j] = f2bf(t2v[j]);
  } else if (i < 393216) {               // t2vT[k][t] = t2v[t][k]
    int j = i - 262144;
    int k = j >> 9, t = j & 511;
    t2vT[j] = f2bf(t2v[t * 256 + k]);
  }
}

// ---- RL = sum(alpha^2) - (sum_t rowsum_t^2)/V ; sigma passthrough ----
extern "C" __global__ void topic_finalize_kernel(const float* __restrict__ tsums,
                                                 const float* __restrict__ sumsq,
                                                 const float* __restrict__ sigma,
                                                 float* __restrict__ outRL,
                                                 float* __restrict__ outSigma)
{
  __shared__ float red[512];
  int t = threadIdx.x;
  float v = tsums[t];
  red[t] = v * v;
  __syncthreads();
  for (int s = 256; s > 0; s >>= 1) {
    if (t < s) red[t] += red[t + s];
    __syncthreads();
  }
  if (t == 0) {
    *outRL = sumsq[0] - red[0] / (float)VOCAB;
    *outSigma = sigma[0];
  }
}

extern "C" void kernel_launch(void* const* d_in, const int* in_sizes, int n_in,
                              void* d_out, int out_size, void* d_ws, size_t ws_size,
                              hipStream_t stream)
{
  (void)in_sizes; (void)n_in; (void)out_size; (void)ws_size;
  const float* w2v   = (const float*)d_in[0];
  const float* t2v   = (const float*)d_in[1];
  const float* A     = (const float*)d_in[2];
  const float* B     = (const float*)d_in[3];
  const float* sigma = (const float*)d_in[4];

  // outputs flat in return order: alpha [512*50000], P [50000], RL [1], s [512*50000], sigma [1]
  float* out       = (float*)d_out;
  float* alpha_out = out;
  float* P_out     = out + (size_t)TOPICS * VOCAB;
  float* RL_out    = P_out + VOCAB;
  float* s_out     = RL_out + 1;
  float* sig_out   = s_out + (size_t)TOPICS * VOCAB;

  char* ws = (char*)d_ws;
  bf16_t* Bt    = (bf16_t*)(ws + 0);        // 128 KB
  bf16_t* Abf   = (bf16_t*)(ws + 131072);   // 128 KB
  bf16_t* t2vbf = (bf16_t*)(ws + 262144);   // 256 KB
  bf16_t* t2vT  = (bf16_t*)(ws + 524288);   // 256 KB
  float*  tsums = (float*)(ws + 786432);    // 512 f32
  float*  sumsq = (float*)(ws + 788480);    // 1 f32

  hipMemsetAsync(ws + 786432, 0, 2048 + 64, stream);

  topic_prep_kernel<<<1536, 256, 0, stream>>>(B, A, t2v, Bt, Abf, t2vbf, t2vT);

  size_t smem = (size_t)4 * WAVE_LDS_BYTES;   // 260 KB / block (CDNA5: 320 KB per WGP)
  hipFuncSetAttribute((const void*)topic_fused_kernel,
                      hipFuncAttributeMaxDynamicSharedMemorySize, (int)smem);
  topic_fused_kernel<<<(VOCAB + 63) / 64, 128, smem, stream>>>(
      w2v, sigma, Bt, Abf, t2vbf, t2vT, alpha_out, s_out, P_out, tsums, sumsq);

  topic_finalize_kernel<<<1, 512, 0, stream>>>(tsums, sumsq, sigma, RL_out, sig_out);
}
